// TransNetNSV2_83133386981989
// MI455X (gfx1250) — matrix operand order repaired and checked by reading, not compiled, not run on hardware
//
#include <hip/hip_runtime.h>
#include <hip/hip_bf16.h>
#include <math.h>

typedef __bf16 bf16;
typedef __attribute__((ext_vector_type(16))) __bf16 v16bf;
typedef __attribute__((ext_vector_type(8)))  __bf16 v8bf;
typedef __attribute__((ext_vector_type(8)))  float  v8f;

#define DEVFN __device__ __forceinline__

// ---------------- problem constants (from reference) ----------------
#define N0_  60000
#define N1_  20000
#define N2_  5000
#define IN_  256
#define HC1_ 256   // H*C1
#define C1_  128
#define HC2_ 128   // H*C2
#define C2_  64
#define E1_  320000
#define E2_  80000

// ---------------- wave reductions (wave32) ----------------
DEVFN float wsum32(float v) {
#pragma unroll
  for (int m = 1; m < 32; m <<= 1) v += __shfl_xor(v, m, 32);
  return v;
}
DEVFN float wmax32(float v) {
#pragma unroll
  for (int m = 1; m < 32; m <<= 1) v = fmaxf(v, __shfl_xor(v, m, 32));
  return v;
}
DEVFN float hsum16(float v) {   // reduce within each 16-lane half
#pragma unroll
  for (int m = 1; m < 16; m <<= 1) v += __shfl_xor(v, m, 32);
  return v;
}

DEVFN void atomicMaxF(float* addr, float val) {
  // classic bit-trick float max (addr must be initialized to -inf)
  if (val >= 0.0f) atomicMax((int*)addr, __float_as_int(val));
  else             atomicMin((unsigned int*)addr, (unsigned int)__float_as_int(val));
}

// ---------------- utility kernels ----------------
__global__ __launch_bounds__(256) void fill_f32(float* p, float v, long n) {
  long i = (long)blockIdx.x * 256 + threadIdx.x;
  if (i < n) p[i] = v;
}

__global__ __launch_bounds__(256) void cvt_f32_bf16(const float* __restrict__ x,
                                                    bf16* __restrict__ y, long n) {
  long i = (long)blockIdx.x * 256 + threadIdx.x;
  if (i < n) y[i] = (bf16)x[i];
}

// W is [K x N] row-major f32; Bt becomes [N x K] row-major bf16
__global__ __launch_bounds__(256) void transpose_cvt(const float* __restrict__ W,
                                                     bf16* __restrict__ Bt,
                                                     int K, int N) {
  int i = blockIdx.x * 256 + threadIdx.x;
  if (i >= K * N) return;
  int k = i / N, n = i % N;
  Bt[(long)n * K + k] = (bf16)W[i];
}

// ---------------- bf16 WMMA GEMM: C[M,N] = A[M,K] @ B[K,N] + bias ----------------
// A: bf16 row-major [M,K]; Bt: bf16 row-major [N,K] (B transposed); C: f32 [M,N].
// One wave computes one 16x16 tile; 8 waves / block. Rows >= Mstore are masked on
// store (A reads may overrun into valid backing rows).
__global__ __launch_bounds__(256)
void gemm_bf16_wmma(const bf16* __restrict__ A, const bf16* __restrict__ Bt,
                    const float* __restrict__ bias, float* __restrict__ C,
                    int N, int K, int Mstore, int tilesN, int tiles) {
  int wave = threadIdx.x >> 5;
  int lane = threadIdx.x & 31;
  int t = blockIdx.x * 8 + wave;
  if (t >= tiles) return;
  int tm = t / tilesN, tn = t % tilesN;
  int m0 = tm << 4, n0 = tn << 4;
  int mh    = lane & 15;   // A row within tile / B column within tile
  int khalf = lane >> 4;   // which K-half this lane holds

  const bf16* pa = A  + (long)(m0 + mh) * K + khalf * 16;
  const bf16* pb = Bt + (long)(n0 + mh) * K + khalf * 16;

  v8f acc = {0.f, 0.f, 0.f, 0.f, 0.f, 0.f, 0.f, 0.f};
  for (int k0 = 0; k0 < K; k0 += 32) {
    union { v16bf v; v8bf h[2]; } a;
    a.h[0] = *(const v8bf*)(pa + k0);        // K = k0+khalf*16  .. +7
    a.h[1] = *(const v8bf*)(pa + k0 + 8);    // K = k0+khalf*16+8.. +15
    v16bf b = *(const v16bf*)(pb + k0);      // K = k0+khalf*16  .. +15, col = n0+mh
    acc = __builtin_amdgcn_wmma_f32_16x16x32_bf16(
        /*neg_a=*/false, a.v, /*neg_b=*/false, b,
        /*c_mod=*/(short)0, acc, /*reuse_a=*/false, /*reuse_b=*/false);
  }

  float badd = bias[n0 + mh];
  int mbase = m0 + khalf * 8;
#pragma unroll
  for (int r = 0; r < 8; ++r) {
    int m = mbase + r;
    if (m < Mstore) C[(long)m * N + n0 + mh] = acc[r] + badd;
  }
}

// ---------------- edge stage ----------------
// One wave per edge; lanes 0-15 -> head 0, lanes 16-31 -> head 1.
__global__ __launch_bounds__(256)
void edge_alpha(const float* __restrict__ q, const float* __restrict__ k,
                const float* __restrict__ We,
                const int* __restrict__ src, const int* __restrict__ dst,
                const int* __restrict__ eid,
                float* __restrict__ alpha, int E, int C, float scale) {
  int e = (blockIdx.x * 256 + threadIdx.x) >> 5;
  if (e >= E) return;
  int lane = threadIdx.x & 31;
  int h = lane >> 4, cl = lane & 15;
  int HC = C << 1;
  int s = src[e], d = dst[e], ed = eid[e];
  const float* qp = q  + (long)d  * HC + h * C;
  const float* kp = k  + (long)s  * HC + h * C;
  const float* ep = We + (long)ed * HC + h * C;
  float acc = 0.f;
  for (int c = cl; c < C; c += 16) acc += qp[c] * (kp[c] + ep[c]);
  acc = hsum16(acc);
  if (cl == 0) alpha[e * 2 + h] = acc * scale;
}

__global__ __launch_bounds__(256)
void edge_amax(const float* __restrict__ alpha, const int* __restrict__ dst,
               float* __restrict__ amax, int E) {
  int i = blockIdx.x * 256 + threadIdx.x;
  if (i >= 2 * E) return;
  int e = i >> 1, h = i & 1;
  atomicMaxF(&amax[(long)dst[e] * 2 + h], alpha[i]);
}

// ex = exp(alpha - amax[dst]); den[dst] += ex; acc[dst] += ex * (v[src] + We[eid])
__global__ __launch_bounds__(256)
void edge_scatter(const float* __restrict__ v, const float* __restrict__ We,
                  const float* __restrict__ alpha, const float* __restrict__ amax,
                  const int* __restrict__ src, const int* __restrict__ dst,
                  const int* __restrict__ eid,
                  float* __restrict__ accum, float* __restrict__ den,
                  int E, int C) {
  int e = (blockIdx.x * 256 + threadIdx.x) >> 5;
  if (e >= E) return;
  int lane = threadIdx.x & 31;
  int h = lane >> 4, cl = lane & 15;
  int HC = C << 1;
  int s = src[e], d = dst[e], ed = eid[e];
  float a  = alpha[e * 2 + h];
  float mx = amax[(long)d * 2 + h];
  float ex = expf(a - mx);
  if (cl == 0) atomicAdd(&den[(long)d * 2 + h], ex);
  const float* vp = v  + (long)s  * HC + h * C;
  const float* ep = We + (long)ed * HC + h * C;
  float* op = accum + (long)d * HC + h * C;
  for (int c = cl; c < C; c += 16) atomicAdd(&op[c], ex * (vp[c] + ep[c]));
}

// ---------------- layer-1 node epilogue ----------------
// out = acc/den (per head); beta-gate with skip; LayerNorm; ReLU; -> bf16 h1
__global__ __launch_bounds__(256)
void node_post1(const float* __restrict__ acc, const float* __restrict__ den,
                const float* __restrict__ xr_in, const float* __restrict__ Wb,
                const float* __restrict__ g, const float* __restrict__ be,
                bf16* __restrict__ h1b, int N) {
  int n = (blockIdx.x * 256 + threadIdx.x) >> 5;
  if (n >= N) return;
  int lane = threadIdx.x & 31;
  float d0 = den[(long)n * 2 + 0] + 1e-16f;
  float d1 = den[(long)n * 2 + 1] + 1e-16f;
  float o[8], xr[8], r[8];
  float p = 0.f;
#pragma unroll
  for (int j = 0; j < 8; ++j) {
    int i = lane + j * 32;                 // 0..255; j<4 -> head 0
    float dv = (j < 4) ? d0 : d1;
    o[j]  = acc[(long)n * 256 + i] / dv;
    xr[j] = xr_in[(long)n * 256 + i];
    p += o[j] * Wb[i] + xr[j] * Wb[256 + i] + (o[j] - xr[j]) * Wb[512 + i];
  }
  p = wsum32(p);
  float beta = 1.f / (1.f + expf(-p));
  float sm = 0.f;
#pragma unroll
  for (int j = 0; j < 8; ++j) { r[j] = beta * xr[j] + (1.f - beta) * o[j]; sm += r[j]; }
  float mean = wsum32(sm) * (1.f / 256.f);
  float vs = 0.f;
#pragma unroll
  for (int j = 0; j < 8; ++j) { float t = r[j] - mean; vs += t * t; }
  float var = wsum32(vs) * (1.f / 256.f);
  float inv = rsqrtf(var + 1e-5f);
#pragma unroll
  for (int j = 0; j < 8; ++j) {
    int i = lane + j * 32;
    float y = (r[j] - mean) * inv * g[i] + be[i];
    y = fmaxf(y, 0.f);
    h1b[(long)n * 256 + i] = (bf16)y;
  }
}

// ---------------- layer-2 node epilogue ----------------
// head-mean; beta-gate; log_softmax over 64 classes -> d_out
__global__ __launch_bounds__(256)
void node_post2(const float* __restrict__ acc, const float* __restrict__ den,
                const float* __restrict__ xr_in, const float* __restrict__ Wb,
                float* __restrict__ out, int N) {
  int n = (blockIdx.x * 256 + threadIdx.x) >> 5;
  if (n >= N) return;
  int lane = threadIdx.x & 31;
  float d0 = den[(long)n * 2 + 0] + 1e-16f;
  float d1 = den[(long)n * 2 + 1] + 1e-16f;
  float o[2], xr[2], r[2];
  float p = 0.f;
#pragma unroll
  for (int j = 0; j < 2; ++j) {
    int c = lane + j * 32;                 // 0..63
    float a0 = acc[(long)n * 128 + c];
    float a1 = acc[(long)n * 128 + 64 + c];
    o[j]  = 0.5f * (a0 / d0 + a1 / d1);
    xr[j] = xr_in[(long)n * 64 + c];
    p += o[j] * Wb[c] + xr[j] * Wb[64 + c] + (o[j] - xr[j]) * Wb[128 + c];
  }
  p = wsum32(p);
  float beta = 1.f / (1.f + expf(-p));
#pragma unroll
  for (int j = 0; j < 2; ++j) r[j] = beta * xr[j] + (1.f - beta) * o[j];
  float mx = wmax32(fmaxf(r[0], r[1]));
  float se = wsum32(expf(r[0] - mx) + expf(r[1] - mx));
  float lse = logf(se);
  out[(long)n * 64 + lane]      = r[0] - mx - lse;
  out[(long)n * 64 + lane + 32] = r[1] - mx - lse;
}

// ---------------- host launcher ----------------
static inline void* wsalloc(char** p, size_t bytes) {
  void* r = *p;
  *p += (bytes + 255) & ~(size_t)255;
  return r;
}

extern "C" void kernel_launch(void* const* d_in, const int* in_sizes, int n_in,
                              void* d_out, int out_size, void* d_ws, size_t ws_size,
                              hipStream_t stream) {
  const float* x   = (const float*)d_in[0];
  const int* src1  = (const int*)d_in[1];
  const int* dst1  = (const int*)d_in[2];
  const int* eid1  = (const int*)d_in[3];
  const int* src2  = (const int*)d_in[4];
  const int* dst2  = (const int*)d_in[5];
  const int* eid2  = (const int*)d_in[6];
  const float* Wq1 = (const float*)d_in[7];  const float* bq1 = (const float*)d_in[8];
  const float* Wk1 = (const float*)d_in[9];  const float* bk1 = (const float*)d_in[10];
  const float* Wv1 = (const float*)d_in[11]; const float* bv1 = (const float*)d_in[12];
  const float* We1 = (const float*)d_in[13];
  const float* Ws1 = (const float*)d_in[14]; const float* bs1 = (const float*)d_in[15];
  const float* Wb1 = (const float*)d_in[16];
  const float* g1  = (const float*)d_in[17]; const float* be1 = (const float*)d_in[18];
  const float* Wq2 = (const float*)d_in[19]; const float* bq2 = (const float*)d_in[20];
  const float* Wk2 = (const float*)d_in[21]; const float* bk2 = (const float*)d_in[22];
  const float* Wv2 = (const float*)d_in[23]; const float* bv2 = (const float*)d_in[24];
  const float* We2 = (const float*)d_in[25];
  const float* Ws2 = (const float*)d_in[26]; const float* bs2 = (const float*)d_in[27];
  const float* Wb2 = (const float*)d_in[28];
  float* out = (float*)d_out;

  // ---- workspace carve-up ----
  char* p = (char*)d_ws;
  bf16* xb    = (bf16*)wsalloc(&p, (size_t)N0_ * IN_ * sizeof(bf16));
  bf16* Wq1t  = (bf16*)wsalloc(&p, (size_t)IN_ * HC1_ * sizeof(bf16));
  bf16* Wk1t  = (bf16*)wsalloc(&p, (size_t)IN_ * HC1_ * sizeof(bf16));
  bf16* Wv1t  = (bf16*)wsalloc(&p, (size_t)IN_ * HC1_ * sizeof(bf16));
  bf16* Ws1t  = (bf16*)wsalloc(&p, (size_t)IN_ * HC1_ * sizeof(bf16));
  bf16* Wq2t  = (bf16*)wsalloc(&p, (size_t)HC1_ * HC2_ * sizeof(bf16));
  bf16* Wk2t  = (bf16*)wsalloc(&p, (size_t)HC1_ * HC2_ * sizeof(bf16));
  bf16* Wv2t  = (bf16*)wsalloc(&p, (size_t)HC1_ * HC2_ * sizeof(bf16));
  bf16* Ws2t  = (bf16*)wsalloc(&p, (size_t)HC1_ * C2_ * sizeof(bf16));
  float* q1   = (float*)wsalloc(&p, (size_t)N1_ * HC1_ * sizeof(float));
  float* k1   = (float*)wsalloc(&p, (size_t)N0_ * HC1_ * sizeof(float));
  float* v1   = (float*)wsalloc(&p, (size_t)N0_ * HC1_ * sizeof(float));
  float* s1   = (float*)wsalloc(&p, (size_t)N1_ * HC1_ * sizeof(float));
  float* al1  = (float*)wsalloc(&p, (size_t)E1_ * 2 * sizeof(float));
  float* am1  = (float*)wsalloc(&p, (size_t)N1_ * 2 * sizeof(float));
  float* dn1  = (float*)wsalloc(&p, (size_t)N1_ * 2 * sizeof(float));
  float* ac1  = (float*)wsalloc(&p, (size_t)N1_ * HC1_ * sizeof(float));
  bf16* h1b   = (bf16*)wsalloc(&p, (size_t)N1_ * HC1_ * sizeof(bf16));
  float* q2   = (float*)wsalloc(&p, (size_t)5008 * HC2_ * sizeof(float));
  float* k2   = (float*)wsalloc(&p, (size_t)N1_ * HC2_ * sizeof(float));
  float* v2   = (float*)wsalloc(&p, (size_t)N1_ * HC2_ * sizeof(float));
  float* s2   = (float*)wsalloc(&p, (size_t)5008 * C2_ * sizeof(float));
  float* al2  = (float*)wsalloc(&p, (size_t)E2_ * 2 * sizeof(float));
  float* am2  = (float*)wsalloc(&p, (size_t)N2_ * 2 * sizeof(float));
  float* dn2  = (float*)wsalloc(&p, (size_t)N2_ * 2 * sizeof(float));
  float* ac2  = (float*)wsalloc(&p, (size_t)N2_ * HC2_ * sizeof(float));

  auto gemm = [&](const bf16* A, const bf16* Bt, const float* bias, float* Cout,
                  int N, int K, int Mstore, int tilesM) {
    int tilesN = N / 16;
    int tiles = tilesM * tilesN;
    int blocks = (tiles + 7) / 8;
    gemm_bf16_wmma<<<blocks, 256, 0, stream>>>(A, Bt, bias, Cout, N, K, Mstore,
                                               tilesN, tiles);
  };
  auto fill = [&](float* ptr, float v, long n) {
    fill_f32<<<(int)((n + 255) / 256), 256, 0, stream>>>(ptr, v, n);
  };

  // ---- precision conversion ----
  cvt_f32_bf16<<<(N0_ * IN_) / 256, 256, 0, stream>>>(x, xb, (long)N0_ * IN_);
  transpose_cvt<<<(IN_ * HC1_ + 255) / 256, 256, 0, stream>>>(Wq1, Wq1t, IN_, HC1_);
  transpose_cvt<<<(IN_ * HC1_ + 255) / 256, 256, 0, stream>>>(Wk1, Wk1t, IN_, HC1_);
  transpose_cvt<<<(IN_ * HC1_ + 255) / 256, 256, 0, stream>>>(Wv1, Wv1t, IN_, HC1_);
  transpose_cvt<<<(IN_ * HC1_ + 255) / 256, 256, 0, stream>>>(Ws1, Ws1t, IN_, HC1_);
  transpose_cvt<<<(HC1_ * HC2_ + 255) / 256, 256, 0, stream>>>(Wq2, Wq2t, HC1_, HC2_);
  transpose_cvt<<<(HC1_ * HC2_ + 255) / 256, 256, 0, stream>>>(Wk2, Wk2t, HC1_, HC2_);
  transpose_cvt<<<(HC1_ * HC2_ + 255) / 256, 256, 0, stream>>>(Wv2, Wv2t, HC1_, HC2_);
  transpose_cvt<<<(HC1_ * C2_ + 255) / 256, 256, 0, stream>>>(Ws2, Ws2t, HC1_, C2_);

  // ---- layer 1 projections (WMMA) ----
  gemm(xb, Wk1t, bk1, k1, HC1_, IN_, N0_, N0_ / 16);
  gemm(xb, Wv1t, bv1, v1, HC1_, IN_, N0_, N0_ / 16);
  gemm(xb, Wq1t, bq1, q1, HC1_, IN_, N1_, N1_ / 16);
  gemm(xb, Ws1t, bs1, s1, HC1_, IN_, N1_, N1_ / 16);

  // ---- layer 1 attention ----
  fill(am1, -INFINITY, (long)N1_ * 2);
  fill(dn1, 0.f, (long)N1_ * 2);
  fill(ac1, 0.f, (long)N1_ * HC1_);
  edge_alpha<<<(E1_ + 7) / 8, 256, 0, stream>>>(q1, k1, We1, src1, dst1, eid1, al1,
                                                E1_, C1_, 0.08838834764831845f);
  edge_amax<<<(2 * E1_ + 255) / 256, 256, 0, stream>>>(al1, dst1, am1, E1_);
  edge_scatter<<<(E1_ + 7) / 8, 256, 0, stream>>>(v1, We1, al1, am1, src1, dst1, eid1,
                                                  ac1, dn1, E1_, C1_);
  node_post1<<<(N1_ + 7) / 8, 256, 0, stream>>>(ac1, dn1, s1, Wb1, g1, be1, h1b, N1_);

  // ---- layer 2 projections (WMMA) ----
  gemm(h1b, Wk2t, bk2, k2, HC2_, HC1_, N1_, N1_ / 16);
  gemm(h1b, Wv2t, bv2, v2, HC2_, HC1_, N1_, N1_ / 16);
  gemm(h1b, Wq2t, bq2, q2, HC2_, HC1_, N2_, (N2_ + 15) / 16);  // masked tail rows
  gemm(h1b, Ws2t, bs2, s2, C2_,  HC1_, N2_, (N2_ + 15) / 16);

  // ---- layer 2 attention ----
  fill(am2, -INFINITY, (long)N2_ * 2);
  fill(dn2, 0.f, (long)N2_ * 2);
  fill(ac2, 0.f, (long)N2_ * HC2_);
  edge_alpha<<<(E2_ + 7) / 8, 256, 0, stream>>>(q2, k2, We2, src2, dst2, eid2, al2,
                                                E2_, C2_, 0.125f);
  edge_amax<<<(2 * E2_ + 255) / 256, 256, 0, stream>>>(al2, dst2, am2, E2_);
  edge_scatter<<<(E2_ + 7) / 8, 256, 0, stream>>>(v2, We2, al2, am2, src2, dst2, eid2,
                                                  ac2, dn2, E2_, C2_);
  node_post2<<<(N2_ + 7) / 8, 256, 0, stream>>>(ac2, dn2, s2, Wb2, out, N2_);
}
